// ExpInstantaneousPhase_69269232550005
// MI455X (gfx1250) — compile-verified, matrix operand order checked by reading, not run on hardware
//
#include <hip/hip_runtime.h>
#include <math.h>

// ---------------------------------------------------------------------------
// Hilbert-phase loss, fused circulant-GEMM on CDNA5 WMMA (gfx1250, wave32)
//
//  hx[t, c] = sum_k taps[(t-k) mod 3000] * x[k, c]     (per shot slab)
//  taps[m]  = (2/n) cot(pi m / n) for odd m, else 0    (n = 3000, even)
//
//  loss = -(1/3000) * sum over all elements of
//         (x/Ax + y/Ay)^2 + (hx/Ax + hy/Ay)^2,  A = sqrt(sig^2 + h^2)
//
//  Precision: split-bf16 (hi+lo) 3-product GEMM ~ fp32-quality accumulation
//  on the bf16 v_wmma_f32_16x16x32_bf16 pipe.
//
//  Round-4 changes:
//   * B planes n-major, stride 20 dwords (16B-aligned, conflict-free):
//     each fragment half is ONE ds_load_b128 straight into fragment VGPRs
//   * tap table pre-shifted by 32 -> branchless circulant gather
// ---------------------------------------------------------------------------

typedef __bf16 bf16;
typedef __attribute__((ext_vector_type(16))) __bf16 v16bf;
typedef __attribute__((ext_vector_type(8)))  float  v8f;
typedef __attribute__((ext_vector_type(4)))  unsigned v4u;

#define NT     3000   // time samples (GEMM M and K)
#define NC     1200   // traces*channels per shot (GEMM N)
#define NSHOT  8
#define KPAD   3008   // NT padded to multiple of 32
#define KTAB   3040   // shifted tap table: ctab[m] = taps[(m-32) mod NT]
#define MTILE  32     // block output rows (2 waves in M)
#define NTILE  128    // block output cols (4 waves x 2 subtiles x 16)
#define KSTEP  32     // bf16 WMMA K
#define KPAIRS (KSTEP / 2)   // 16 K-pairs per step
#define BSTR   20     // B plane row stride in dwords (16B aligned, padded)

// fragment register view: dword j = bf16 elements (2j, 2j+1)
union FRAG16 {
  v16bf    v;
  v4u      q[2];
  unsigned u[8];
};

__device__ __forceinline__ unsigned bfbits(float v) {
  return (unsigned)__builtin_bit_cast(unsigned short, (bf16)v);
}
__device__ __forceinline__ float bflo(float v) {
  bf16 h = (bf16)v;
  return v - (float)h;
}
// packed tap word: {hi part : low16, lo part : high16}
__device__ __forceinline__ unsigned pack_split(float v) {
  return bfbits(v) | (bfbits(bflo(v)) << 16);
}

#define WMMA_BF16(A, B, C) \
  __builtin_amdgcn_wmma_f32_16x16x32_bf16(false, (A), false, (B), (short)0, (C), false, false)

// ---------------------------------------------------------------------------
// Kernel 1: shifted packed split-bf16 Hilbert taps (12 KB)
// ---------------------------------------------------------------------------
__global__ void build_taps_kernel(unsigned* __restrict__ ctab) {
  int m = blockIdx.x * blockDim.x + threadIdx.x;
  if (m >= KTAB) return;
  int mm = (m - 32) % NT;
  if (mm < 0) mm += NT;
  float c = 0.0f;
  if (mm & 1) {
    float ang = 3.14159265358979323846f * (float)mm / (float)NT;
    c = (2.0f / (float)NT) / tanf(ang);   // (2/n) cot(pi m / n), odd m only
  }
  ctab[m] = pack_split(c);
}

// ---------------------------------------------------------------------------
// Kernel 2: fused circulant-GEMM (x and y) + instantaneous-phase loss reduce
// ---------------------------------------------------------------------------
__global__ __launch_bounds__(256)
void hilbert_phase_loss_kernel(const float* __restrict__ x,
                               const float* __restrict__ y,
                               const unsigned* __restrict__ ctab,
                               float* __restrict__ out) {
  __shared__ __align__(16) unsigned s_tap[KTAB];          // shifted taps
  __shared__ __align__(16) unsigned sBxh[NTILE * BSTR];   // x hi plane (10 KB)
  __shared__ __align__(16) unsigned sBxl[NTILE * BSTR];   // x lo plane
  __shared__ __align__(16) unsigned sByh[NTILE * BSTR];   // y hi plane
  __shared__ __align__(16) unsigned sByl[NTILE * BSTR];   // y lo plane
  __shared__ float s_red[8];

  const int tid  = threadIdx.x;
  const int lane = tid & 31;                   // wave32
  const int wave = tid >> 5;                   // 8 waves: 2 (M) x 4 (N)
  const int lo16 = lane & 15;
  const int hih  = lane >> 4;                  // lane half selects K sub-range
  const int wm   = wave >> 2;                  // 0..1
  const int wn   = wave & 3;                   // 0..3

  const int n0   = blockIdx.x * NTILE;         // column tile
  const int m0   = blockIdx.y * MTILE;         // output-time tile
  const int shot = blockIdx.z;
  const long long slab = (long long)shot * NT * NC;

  // cache shifted tap table in LDS
  for (int m = tid; m < KTAB; m += 256) s_tap[m] = ctab[m];

  v8f accx[2] = {{}, {}};                      // hx accumulators (2 N-subtiles)
  v8f accy[2] = {{}, {}};                      // hy accumulators

  const int trow  = m0 + wm * 16 + lo16;       // A-fragment output-time row
  // checkerboard: nonzero tap sits in the (trow&1 ? even-K : odd-K) element
  const int shamt = (trow & 1) ? 0 : 16;       // place into low/high half
  int base = (trow >= NT) ? trow - NT : trow;  // (trow - k0) mod NT

  for (int k0 = 0; k0 < KPAD; k0 += KSTEP) {
    __syncthreads();                           // previous-iter LDS reads done
    // ---- stage B planes: packed K-pair word per (n, kp), split once ----
    for (int e = tid; e < KPAIRS * NTILE; e += 256) {
      int kp = e >> 7;                         // K-pair index 0..15
      int nn = e & 127;
      int kg = k0 + 2 * kp, ng = n0 + nn;
      float x0 = 0.f, x1 = 0.f, y0 = 0.f, y1 = 0.f;
      if (ng < NC) {
        long long off = slab + (long long)kg * NC + ng;
        if (kg < NT)     { x0 = x[off];      y0 = y[off];      }
        if (kg + 1 < NT) { x1 = x[off + NC]; y1 = y[off + NC]; }
      }
      int w = nn * BSTR + kp;                  // n-major, padded stride
      sBxh[w] = bfbits(x0) | (bfbits(x1) << 16);
      sBxl[w] = bfbits(bflo(x0)) | (bfbits(bflo(x1)) << 16);
      sByh[w] = bfbits(y0) | (bfbits(y1) << 16);
      sByl[w] = bfbits(bflo(y0)) | (bfbits(bflo(y1)) << 16);
    }
    __syncthreads();

    // ---- A fragments: branchless gather, one b32 tap load per K-pair ----
    FRAG16 ahi, alo;
    {
      // shifted table: +32 bias keeps every index in [1, KTAB-1]
      const int b2 = base + 32 - ((trow & 1) ? 0 : 1) - 8 * hih;
#pragma unroll
      for (int j = 0; j < 8; ++j) {
        unsigned p = s_tap[b2 - 2 * j - ((j >= 4) ? 8 : 0)];
        ahi.u[j] = (p & 0xffffu) << shamt;     // other half stays zero
        alo.u[j] = (p >> 16) << shamt;
      }
    }

    // ---- two N-subtiles reuse the same A fragments ----
#pragma unroll
    for (int s = 0; s < 2; ++s) {
      const int rowb = (wn * 16 + s * 64 + lo16) * BSTR;
      const int h0 = rowb + 4 * hih;           // kp = 4*hih + (0..3)
      const int h1 = rowb + 8 + 4 * hih;       // kp = 8 + 4*hih + (0..3)
      FRAG16 bxh, bxl, byh, byl;
      bxh.q[0] = *(const v4u*)&sBxh[h0];       // one ds_load_b128 per half,
      bxh.q[1] = *(const v4u*)&sBxh[h1];       // straight into fragment regs
      bxl.q[0] = *(const v4u*)&sBxl[h0];
      bxl.q[1] = *(const v4u*)&sBxl[h1];
      byh.q[0] = *(const v4u*)&sByh[h0];
      byh.q[1] = *(const v4u*)&sByh[h1];
      byl.q[0] = *(const v4u*)&sByl[h0];
      byl.q[1] = *(const v4u*)&sByl[h1];
      // 6 WMMAs, x/y interleaved (independent accumulator chains)
      accx[s] = WMMA_BF16(ahi.v, bxh.v, accx[s]);
      accy[s] = WMMA_BF16(ahi.v, byh.v, accy[s]);
      accx[s] = WMMA_BF16(ahi.v, bxl.v, accx[s]);
      accy[s] = WMMA_BF16(ahi.v, byl.v, accy[s]);
      accx[s] = WMMA_BF16(alo.v, bxh.v, accx[s]);
      accy[s] = WMMA_BF16(alo.v, byh.v, accy[s]);
    }

    base -= KSTEP;                             // maintain (trow - k0) mod NT
    if (base < 0) base += NT;
  }

  // ---- epilogue: loss terms at this tile, then block reduction ----
  // C/D layout: lane -> col = lo16 (+ half selects M+8); VGPR r -> row r (+8)
  float lsum = 0.0f;
#pragma unroll
  for (int s = 0; s < 2; ++s) {
    const int ngl = n0 + wn * 16 + s * 64 + lo16;
#pragma unroll
    for (int r = 0; r < 8; ++r) {
      int tg = m0 + wm * 16 + hih * 8 + r;
      if (tg < NT && ngl < NC) {
        long long off = slab + (long long)tg * NC + ngl;
        float xv = x[off], yv = y[off];
        float hx = accx[s][r], hy = accy[s][r];
        float invAx = rsqrtf(xv * xv + hx * hx);
        float invAy = rsqrtf(yv * yv + hy * hy);
        float p1 = xv * invAx + yv * invAy;    // cos(phi_x) + cos(phi_y)
        float p2 = hx * invAx + hy * invAy;    // sin(phi_x) + sin(phi_y)
        lsum += p1 * p1 + p2 * p2;
      }
    }
  }
  // wave32 butterfly reduce
#pragma unroll
  for (int o = 16; o > 0; o >>= 1) lsum += __shfl_xor(lsum, o, 32);
  if (lane == 0) s_red[wave] = lsum;
  __syncthreads();
  if (tid == 0) {
    float b = 0.0f;
#pragma unroll
    for (int w = 0; w < 8; ++w) b += s_red[w];
    atomicAdd(out, -b / (float)NT);            // loss = -sum(mean over time)
  }
}

// ---------------------------------------------------------------------------
extern "C" void kernel_launch(void* const* d_in, const int* in_sizes, int n_in,
                              void* d_out, int out_size, void* d_ws, size_t ws_size,
                              hipStream_t stream) {
  const float* x = (const float*)d_in[0];
  const float* y = (const float*)d_in[1];
  float* out = (float*)d_out;

  // workspace: shifted packed tap table (3040 x 4 bytes)
  unsigned* ctab = (unsigned*)d_ws;

  hipMemsetAsync(d_out, 0, sizeof(float), stream);  // graph-capturable
  build_taps_kernel<<<(KTAB + 255) / 256, 256, 0, stream>>>(ctab);

  dim3 grid((NC + NTILE - 1) / NTILE,   // 10 column tiles
            (NT + MTILE - 1) / MTILE,   // 94 output-time tiles
            NSHOT);                     // 8 shots
  hilbert_phase_loss_kernel<<<grid, 256, 0, stream>>>(x, y, ctab, out);
}
